// MetaAtt_17566416241060
// MI455X (gfx1250) — matrix operand order repaired and checked by reading, not compiled, not run on hardware
//
#include <hip/hip_runtime.h>
#include <hip/hip_fp16.h>

#define N_NODES 8192
#define D_INF   256
#define D_OUTF  64
#define H_HEADS 4
#define ALPHA   0.2f
#define LOG2E   1.4426950408889634f

typedef __attribute__((ext_vector_type(16))) _Float16 v16h;
typedef __attribute__((ext_vector_type(8)))  _Float16 v8h;
typedef __attribute__((ext_vector_type(8)))  float    v8f;

__device__ __forceinline__ float fast_exp2(float x) {
#if __has_builtin(__builtin_amdgcn_exp2f)
    return __builtin_amdgcn_exp2f(x);        // v_exp_f32
#else
    return __expf(x * 0.6931471805599453f);  // exp(x*ln2) == 2^x
#endif
}

__device__ __forceinline__ unsigned enc_ord(float x) {
    unsigned b = __float_as_uint(x);
    return (b & 0x80000000u) ? ~b : (b | 0x80000000u);
}
__device__ __forceinline__ float dec_ord(unsigned e) {
    unsigned b = (e & 0x80000000u) ? (e & 0x7FFFFFFFu) : ~e;
    return __uint_as_float(b);
}

// ---------------------------------------------------------------------------
// Kernel 0: zero the atomic accumulators (ws is poisoned, must init each call)
// ---------------------------------------------------------------------------
__global__ void gat_init_kernel(unsigned* __restrict__ dmax_enc,
                                float* __restrict__ sumWh) {
    int t = threadIdx.x;
    if (t < H_HEADS) dmax_enc[t] = 0u;                 // below any real encoding
    if (t < H_HEADS * D_OUTF) sumWh[t] = 0.0f;
}

// ---------------------------------------------------------------------------
// Kernel 1: Wh = x @ W[h]  (f32, tiled LDS GEMM), plus
//   WhT_f16[h][c][i]  (transposed f16 copy, B-operand friendly)
//   f_src/f_dst pre-scaled by log2(e) (internal only), dmax_enc, sumWh
// ---------------------------------------------------------------------------
__global__ void __launch_bounds__(256)
gat_wh_kernel(const float* __restrict__ x,
              const float* __restrict__ W,
              const float* __restrict__ a_src,
              const float* __restrict__ a_dst,
              _Float16* __restrict__ WhT,
              float* __restrict__ f_src,
              float* __restrict__ f_dst,
              unsigned* __restrict__ dmax_enc,
              float* __restrict__ sumWh)
{
    const int h   = blockIdx.y;
    const int i0  = blockIdx.x * 64;
    const int t   = threadIdx.x;
    const int row = t >> 2;      // 0..63
    const int cg  = t & 3;       // column group: 16 cols each

    __shared__ float sX[64 * 65];
    __shared__ float sW[64 * 65];
    __shared__ float ssum[D_OUTF];
    if (t < D_OUTF) ssum[t] = 0.0f;

    float acc[16];
#pragma unroll
    for (int cc = 0; cc < 16; ++cc) acc[cc] = 0.0f;

    for (int k0 = 0; k0 < D_INF; k0 += 64) {
        __syncthreads();
        for (int idx = t; idx < 64 * 64; idx += 256) {
            int r = idx >> 6, k = idx & 63;
            sX[r * 65 + k] = x[(size_t)(i0 + r) * D_INF + k0 + k];
        }
        for (int idx = t; idx < 64 * 64; idx += 256) {
            int k = idx >> 6, c = idx & 63;
            sW[k * 65 + c] = W[((size_t)h * D_INF + k0 + k) * D_OUTF + c];
        }
        __syncthreads();
        for (int k = 0; k < 64; ++k) {
            float xv = sX[row * 65 + k];
#pragma unroll
            for (int cc = 0; cc < 16; ++cc)
                acc[cc] = fmaf(xv, sW[k * 65 + cg * 16 + cc], acc[cc]);
        }
    }

    const int i = i0 + row;
    float fs = 0.0f, fd = 0.0f;
#pragma unroll
    for (int cc = 0; cc < 16; ++cc) {
        int c = cg * 16 + cc;
        WhT[(size_t)(h * D_OUTF + c) * N_NODES + i] = (_Float16)acc[cc];
        fs = fmaf(acc[cc], a_src[h * D_OUTF + c], fs);
        fd = fmaf(acc[cc], a_dst[h * D_OUTF + c], fd);
        atomicAdd(&ssum[c], acc[cc]);
    }
    // 4 threads (lanes t, t^1, t^2, t^3 within one wave) share a row
    fs += __shfl_xor(fs, 1, 32);  fs += __shfl_xor(fs, 2, 32);
    fd += __shfl_xor(fd, 1, 32);  fd += __shfl_xor(fd, 2, 32);
    if (cg == 0) {
        // store in log2 domain: softmax(exp(lr)) == softmax(exp2(lr*log2e))
        float fs2 = fs * LOG2E, fd2 = fd * LOG2E;
        f_src[h * N_NODES + i] = fs2;
        f_dst[h * N_NODES + i] = fd2;
        atomicMax(&dmax_enc[h], enc_ord(fd2));
    }
    __syncthreads();
    if (t < D_OUTF) atomicAdd(&sumWh[h * D_OUTF + t], ssum[t]);
}

// ---------------------------------------------------------------------------
// Kernel 2: fused masked-softmax + att@Wh via V_WMMA_F32_16X16X32_F16.
// Block = 128 threads = 4 waves, one wave per head, all over the same 16 rows
// (adj tile shared across heads through the WGP cache).
// ---------------------------------------------------------------------------
__global__ void __launch_bounds__(128)
gat_attention_kernel(const int* __restrict__ adj,
                     const float* __restrict__ f_src,
                     const float* __restrict__ f_dst,
                     const _Float16* __restrict__ WhT,
                     const unsigned* __restrict__ dmax_enc,
                     const float* __restrict__ sumWh,
                     float* __restrict__ out)
{
    const int h  = threadIdx.x >> 5;   // head = wave id
    const int ln = threadIdx.x & 31;   // lane in wave32
    const int r  = ln & 15;
    const int hi = ln >> 4;
    const int i0 = blockIdx.x * 16;
    const int i  = i0 + r;

    const float s     = f_src[h * N_NODES + i];        // already * log2(e)
    const float dmax  = dec_ord(dmax_enc[h]);
    const float sd    = s + dmax;
    const float shift = fmaxf(sd, ALPHA * sd);         // LR upper bound, log2 domain
    // LR(s+d) - shift == max((s-shift) + d, (a*s-shift) + a*d): fold per-lane
    const float p = s - shift;
    const float q = ALPHA * s - shift;

    // A-operand K ownership (16-bit A 16x32 layout): lanes<16 own K in
    // {0..7,16..23}; lanes>=16 own K in {8..15,24..31}.
    const int kbase = hi ? 8 : 0;
    const int*   arow = adj   + (size_t)i * N_NODES + kbase;
    const float* drow = f_dst + (size_t)h * N_NODES + kbase;

    v8f acc[4];
#pragma unroll
    for (int tt = 0; tt < 4; ++tt) {
        v8f z = {0.f, 0.f, 0.f, 0.f, 0.f, 0.f, 0.f, 0.f};
        acc[tt] = z;
    }
    float zp0 = 0.0f, zp1 = 0.0f, zp2 = 0.0f, zp3 = 0.0f;

    for (int j0 = 0; j0 < N_NODES; j0 += 32) {
        __builtin_prefetch(arow + j0 + 64, 0, 3);   // speculative: safe past end

        int4 a0 = *(const int4*)(arow + j0);
        int4 a1 = *(const int4*)(arow + j0 + 4);
        int4 a2 = *(const int4*)(arow + j0 + 16);
        int4 a3 = *(const int4*)(arow + j0 + 20);
        float4 d0 = *(const float4*)(drow + j0);
        float4 d1 = *(const float4*)(drow + j0 + 4);
        float4 d2 = *(const float4*)(drow + j0 + 16);
        float4 d3 = *(const float4*)(drow + j0 + 20);

        int av[16]   = {a0.x, a0.y, a0.z, a0.w, a1.x, a1.y, a1.z, a1.w,
                        a2.x, a2.y, a2.z, a2.w, a3.x, a3.y, a3.z, a3.w};
        float dv[16] = {d0.x, d0.y, d0.z, d0.w, d1.x, d1.y, d1.z, d1.w,
                        d2.x, d2.y, d2.z, d2.w, d3.x, d3.y, d3.z, d3.w};

        v16h A;
        float wv[16];
#pragma unroll
        for (int n = 0; n < 16; ++n) {
            float d   = dv[n];
            float lrs = fmaxf(p + d, fmaf(ALPHA, d, q));        // LR(s+d)-shift
            float arg = (av[n] > 0) ? lrs : -1.0e30f;           // cndmask, no branch
            float w   = fast_exp2(arg);                         // v_exp_f32
            wv[n] = w;
            A[n]  = (_Float16)w;
        }
        // 4 parallel partial sums: short dependency chains under the WMMAs
#pragma unroll
        for (int n = 0; n < 16; n += 4) {
            zp0 += wv[n + 0];
            zp1 += wv[n + 1];
            zp2 += wv[n + 2];
            zp3 += wv[n + 3];
        }

        // B: WhT[h][col][j] — lane owns column (ln&15), K-run picked by `hi`.
#pragma unroll
        for (int tt = 0; tt < 4; ++tt) {
            const _Float16* bp =
                WhT + (size_t)(h * D_OUTF + tt * 16 + r) * N_NODES + j0 + hi * 8;
            v16h B;
            ((v8h*)&B)[0] = *(const v8h*)(bp);
            ((v8h*)&B)[1] = *(const v8h*)(bp + 16);
            acc[tt] = __builtin_amdgcn_wmma_f32_16x16x32_f16(
                false, A, false, B, (short)0, acc[tt], false, false);
        }
    }

    float zpart = (zp0 + zp1) + (zp2 + zp3);
    // Row softmax denominator: each row's 32 K-columns are split across lane
    // pair (ln, ln^16); combine, then broadcast per row through LDS.
    zpart += __shfl_xor(zpart, 16, 32);
    __shared__ float zb[H_HEADS * 16];
    if (ln < 16) zb[h * 16 + r] = zpart;
    __syncthreads();

    float izrow[8], fbflag[8];
#pragma unroll
    for (int v = 0; v < 8; ++v) {
        float z = zb[h * 16 + v + 8 * hi];     // C-layout row of VGPR v
        fbflag[v] = (z > 0.0f) ? 0.0f : 1.0f;
        izrow[v]  = (z > 0.0f) ? 1.0f / z : 0.0f;
    }

#pragma unroll
    for (int tt = 0; tt < 4; ++tt) {
        int c = tt * 16 + r;
        float fb = sumWh[h * D_OUTF + c] * (1.0f / (float)N_NODES); // all-masked row
#pragma unroll
        for (int v = 0; v < 8; ++v) {
            int rowi = i0 + v + 8 * hi;
            float val = acc[tt][v] * izrow[v] + fbflag[v] * fb;
            out[(size_t)rowi * (H_HEADS * D_OUTF) + h * D_OUTF + c] = val;
        }
    }
}

// ---------------------------------------------------------------------------
extern "C" void kernel_launch(void* const* d_in, const int* in_sizes, int n_in,
                              void* d_out, int out_size, void* d_ws, size_t ws_size,
                              hipStream_t stream)
{
    const float* x     = (const float*)d_in[0];
    const int*   adj   = (const int*)d_in[1];
    const float* W     = (const float*)d_in[2];
    const float* a_src = (const float*)d_in[3];
    const float* a_dst = (const float*)d_in[4];
    float* out = (float*)d_out;

    char* ws = (char*)d_ws;
    _Float16* WhT = (_Float16*)ws;
    ws += (size_t)H_HEADS * D_OUTF * N_NODES * sizeof(_Float16);   // 4 MB
    float* f_src = (float*)ws;  ws += (size_t)H_HEADS * N_NODES * sizeof(float);
    float* f_dst = (float*)ws;  ws += (size_t)H_HEADS * N_NODES * sizeof(float);
    float* sumWh = (float*)ws;  ws += (size_t)H_HEADS * D_OUTF * sizeof(float);
    unsigned* dmax_enc = (unsigned*)ws;

    gat_init_kernel<<<1, 256, 0, stream>>>(dmax_enc, sumWh);
    gat_wh_kernel<<<dim3(N_NODES / 64, H_HEADS), 256, 0, stream>>>(
        x, W, a_src, a_dst, WhT, f_src, f_dst, dmax_enc, sumWh);
    gat_attention_kernel<<<N_NODES / 16, 128, 0, stream>>>(
        adj, f_src, f_dst, WhT, dmax_enc, sumWh, out);
}
